// CrossAttentionLayer_19129784336817
// MI455X (gfx1250) — matrix operand order repaired and checked by reading, not compiled, not run on hardware
//
#include <hip/hip_runtime.h>
#include <hip/hip_bf16.h>

// ---------------------------------------------------------------------------
// CrossAttentionLayer for MI455X (gfx1250, wave32, WMMA bf16 f32-accum)
// ---------------------------------------------------------------------------
typedef __bf16 bf16;
typedef __attribute__((ext_vector_type(8)))  __bf16 v8bf;
typedef __attribute__((ext_vector_type(16))) __bf16 v16bf;
typedef __attribute__((ext_vector_type(8)))  float  v8f;

static constexpr int BB   = 2;     // batch
static constexpr int N1   = 2048;  // query length
static constexpr int N2   = 2048;  // key length
static constexpr int EMB  = 1024;
static constexpr int QD   = 512;   // q/k/v input dim
static constexpr int NH   = 16;
static constexpr int HD   = 64;    // head dim

__device__ __forceinline__ bf16 f2bf(float f) {
    unsigned u = __builtin_bit_cast(unsigned, f);
    u = u + 0x7FFFu + ((u >> 16) & 1u);          // round-to-nearest-even
    unsigned short s = (unsigned short)(u >> 16);
    return __builtin_bit_cast(bf16, s);
}

// ---- DPP16 butterfly reductions within 16-lane rows (no LDS, no waits) ----
template <int CTRL>
__device__ __forceinline__ float dppf(float x) {
    int i = __builtin_bit_cast(int, x);
    int r = __builtin_amdgcn_update_dpp(i, i, CTRL, 0xF, 0xF, true);
    return __builtin_bit_cast(float, r);
}
__device__ __forceinline__ float red_max16(float x) {
    x = fmaxf(x, dppf<0xB1>(x));    // quad_perm(1,0,3,2)  : xor 1
    x = fmaxf(x, dppf<0x4E>(x));    // quad_perm(2,3,0,1)  : xor 2
    x = fmaxf(x, dppf<0x141>(x));   // row_half_mirror     : combine quads
    x = fmaxf(x, dppf<0x140>(x));   // row_mirror          : combine octets
    return x;
}
__device__ __forceinline__ float red_sum16(float x) {
    x += dppf<0xB1>(x);
    x += dppf<0x4E>(x);
    x += dppf<0x141>(x);
    x += dppf<0x140>(x);
    return x;
}

// A-fragment (16x32 bf16, MxK): lane m holds row m; dwords 0-3 = K 8*hi..8*hi+7,
// dwords 4-7 = K 16+8*hi..23+8*hi  (ISA 7.12.2). Two contiguous 16B loads.
__device__ __forceinline__ v16bf load_frag_a(const bf16* rowbase, int hi) {
    v8bf lo  = *(const v8bf*)(rowbase + 8 * hi);
    v8bf hiv = *(const v8bf*)(rowbase + 16 + 8 * hi);
    v16bf r;
#pragma unroll
    for (int i = 0; i < 8; ++i) { r[i] = lo[i]; r[8 + i] = hiv[i]; }
    return r;
}

// B-fragment (32x16 bf16, KxN): lane n holds column n; elems 0-7 = K 8*hi..8*hi+7,
// elems 8-15 = K 16+8*hi..23+8*hi. colbase points at (k0, n) with K contiguous.
__device__ __forceinline__ v16bf load_frag_b(const bf16* colbase, int hi) {
    v8bf lo  = *(const v8bf*)(colbase + 8 * hi);
    v8bf hiv = *(const v8bf*)(colbase + 16 + 8 * hi);
    v16bf r;
#pragma unroll
    for (int i = 0; i < 8; ++i) { r[i] = lo[i]; r[8 + i] = hiv[i]; }
    return r;
}

// ---------------------------------------------------------------------------
// Elementwise f32 -> bf16
__global__ void cvt_f32_bf16_kernel(const float* __restrict__ in,
                                    bf16* __restrict__ out, int n) {
    for (int i = blockIdx.x * blockDim.x + threadIdx.x; i < n;
         i += gridDim.x * blockDim.x)
        out[i] = f2bf(in[i]);
}

// W[K,N] (f32) -> Wt[N,K] (bf16)
__global__ void cvt_transpose_kernel(const float* __restrict__ W,
                                     bf16* __restrict__ Wt, int K, int N) {
    int total = K * N;
    for (int i = blockIdx.x * blockDim.x + threadIdx.x; i < total;
         i += gridDim.x * blockDim.x) {
        int k = i / N, n = i % N;
        Wt[(size_t)n * K + k] = f2bf(W[i]);
    }
}

// Vproj[B*N2, EMB] (bf16) -> Vt[(b*NH+h)*HD + d][N2] (bf16)
__global__ void v_transpose_kernel(const bf16* __restrict__ Vp,
                                   bf16* __restrict__ Vt) {
    int total = BB * N2 * EMB;
    for (int i = blockIdx.x * blockDim.x + threadIdx.x; i < total;
         i += gridDim.x * blockDim.x) {
        int d0  = i % EMB;
        int row = i / EMB;            // b*N2 + n
        int n   = row % N2;
        int b   = row / N2;
        int h   = d0 >> 6;
        int d   = d0 & 63;
        Vt[((size_t)((b * NH + h) * HD + d)) * N2 + n] = Vp[i];
    }
}

// ---------------------------------------------------------------------------
// C[M,N] = A[M,K] @ Bt[N,K]^T + bias   (bf16 in, f32 accum via WMMA)
// One wave computes a 32x64 tile (2 row-groups x 4 col-groups): each
// B-fragment feeds two WMMAs. grid = (N/64, M/32), block = 32.
template <typename OutT>
__global__ __launch_bounds__(32, 1)
void gemm_bias_wmma_kernel(const bf16* __restrict__ A,
                           const bf16* __restrict__ Bt,
                           const float* __restrict__ bias,
                           OutT* __restrict__ C,
                           int M, int N, int K) {
    int lane = threadIdx.x;
    int m    = lane & 15;
    int hi   = lane >> 4;
    int tile_m = blockIdx.y * 32;
    int n_base = blockIdx.x * 64;

    const bf16* arow0 = A + (size_t)(tile_m + m) * K;
    const bf16* arow1 = A + (size_t)(tile_m + 16 + m) * K;
    v8f acc[2][4];
#pragma unroll
    for (int r = 0; r < 2; ++r)
#pragma unroll
        for (int c = 0; c < 4; ++c)
#pragma unroll
            for (int j = 0; j < 8; ++j) acc[r][c][j] = 0.0f;

    for (int k0 = 0; k0 < K; k0 += 32) {
        v16bf a0 = load_frag_a(arow0 + k0, hi);
        v16bf a1 = load_frag_a(arow1 + k0, hi);
#pragma unroll
        for (int c = 0; c < 4; ++c) {
            const bf16* brow = Bt + (size_t)(n_base + c * 16 + m) * K + k0;
            v16bf b = load_frag_b(brow, hi);
            acc[0][c] = __builtin_amdgcn_wmma_f32_16x16x32_bf16(
                false, a0, false, b, (short)0, acc[0][c], false, false);
            acc[1][c] = __builtin_amdgcn_wmma_f32_16x16x32_bf16(
                false, a1, false, b, (short)0, acc[1][c], false, false);
        }
    }

#pragma unroll
    for (int r = 0; r < 2; ++r)
#pragma unroll
    for (int c = 0; c < 4; ++c) {
        int col  = n_base + c * 16 + m;
        float bv = bias[col];
#pragma unroll
        for (int j = 0; j < 8; ++j) {
            int row = tile_m + r * 16 + j + 8 * hi;
            float v = acc[r][c][j] + bv;
            if constexpr (__is_same(OutT, float))
                C[(size_t)row * N + col] = v;
            else
                C[(size_t)row * N + col] = f2bf(v);
        }
    }
}

// ---------------------------------------------------------------------------
// Flash attention: one wave per (b, h, 16-row query tile); streams 32 keys
// per iteration, online softmax (DPP16 reductions), P bounced through LDS
// into WMMA A-layout.
__global__ __launch_bounds__(32, 1)
void attn_flash_kernel(const bf16* __restrict__ Qp,  // [B*N1, EMB]
                       const bf16* __restrict__ Kp,  // [B*N2, EMB]
                       const bf16* __restrict__ Vt,  // [B*NH*HD, N2]
                       bf16* __restrict__ Op) {      // [B*N1, EMB]
    __shared__ bf16 ldsP[16 * 32];

    int lane = threadIdx.x;
    int m    = lane & 15;
    int hi   = lane >> 4;
    int qt   = blockIdx.x;   // query tile (0..N1/16-1)
    int h    = blockIdx.y;
    int b    = blockIdx.z;

    const float scale = 0.125f;  // HEAD_DIM^-0.5

    const bf16* qrow = Qp + (size_t)(b * N1 + qt * 16 + m) * EMB + h * HD;
    v16bf q0 = load_frag_a(qrow, hi);        // head dims 0..31
    v16bf q1 = load_frag_a(qrow + 32, hi);   // head dims 32..63

    v8f o[4];
    float mrun[8], lrun[8];
#pragma unroll
    for (int c = 0; c < 4; ++c)
#pragma unroll
        for (int j = 0; j < 8; ++j) o[c][j] = 0.0f;
#pragma unroll
    for (int j = 0; j < 8; ++j) { mrun[j] = -1e30f; lrun[j] = 0.0f; }

    const bf16* krow0 = Kp + (size_t)(b * N2 + m) * EMB + h * HD;
    const bf16* krow1 = Kp + (size_t)(b * N2 + 16 + m) * EMB + h * HD;

    for (int kt = 0; kt < N2; kt += 32) {
        // prefetch next key tile (64 KB ahead) while we do this tile's math
        __builtin_prefetch((const void*)(krow0 + 32 * EMB), 0, 1);
        __builtin_prefetch((const void*)(krow1 + 32 * EMB), 0, 1);

        // ---- scores S = Q * K^T for 32 keys (two 16-key C tiles) ----
        v8f s0, s1;
#pragma unroll
        for (int j = 0; j < 8; ++j) { s0[j] = 0.0f; s1[j] = 0.0f; }

        v16bf kb;
        kb = load_frag_b(krow0, hi);
        s0 = __builtin_amdgcn_wmma_f32_16x16x32_bf16(false, q0, false, kb,
                                                     (short)0, s0, false, false);
        kb = load_frag_b(krow0 + 32, hi);
        s0 = __builtin_amdgcn_wmma_f32_16x16x32_bf16(false, q1, false, kb,
                                                     (short)0, s0, false, false);
        kb = load_frag_b(krow1, hi);
        s1 = __builtin_amdgcn_wmma_f32_16x16x32_bf16(false, q0, false, kb,
                                                     (short)0, s1, false, false);
        kb = load_frag_b(krow1 + 32, hi);
        s1 = __builtin_amdgcn_wmma_f32_16x16x32_bf16(false, q1, false, kb,
                                                     (short)0, s1, false, false);

        // ---- online softmax update (row j lives in C-VGPR j, 16 lanes) ----
#pragma unroll
        for (int j = 0; j < 8; ++j) {
            float a0 = s0[j] * scale;
            float a1 = s1[j] * scale;
            float mx   = red_max16(fmaxf(a0, a1));
            float mnew = fmaxf(mrun[j], mx);
            float corr = __expf(mrun[j] - mnew);
            float p0 = __expf(a0 - mnew);
            float p1 = __expf(a1 - mnew);
            float rs = red_sum16(p0 + p1);
            lrun[j] = lrun[j] * corr + rs;
            mrun[j] = mnew;
            o[0][j] *= corr; o[1][j] *= corr;
            o[2][j] *= corr; o[3][j] *= corr;

            int row = j + 8 * hi;                // actual tile row of this lane
            ldsP[row * 32 + m]      = f2bf(p0);  // keys kt..kt+15
            ldsP[row * 32 + 16 + m] = f2bf(p1);  // keys kt+16..kt+31
        }

        // same-wave DS store -> DS load ordering
        asm volatile("s_wait_dscnt 0" ::: "memory");

        // P as A-fragment (16 rows x 32 keys)
        v16bf p = load_frag_a(&ldsP[m * 32], hi);

        // ---- O += P @ V (4 chunks of 16 head-dim columns) ----
#pragma unroll
        for (int c = 0; c < 4; ++c) {
            const bf16* vrow =
                Vt + ((size_t)((b * NH + h) * HD + c * 16 + m)) * N2 + kt;
            v16bf vb = load_frag_b(vrow, hi);
            o[c] = __builtin_amdgcn_wmma_f32_16x16x32_bf16(
                false, p, false, vb, (short)0, o[c], false, false);
        }

        krow0 += 32 * EMB;
        krow1 += 32 * EMB;
    }

    // ---- normalize and store ----
#pragma unroll
    for (int j = 0; j < 8; ++j) {
        float inv = 1.0f / lrun[j];
        int row = qt * 16 + j + 8 * hi;
#pragma unroll
        for (int c = 0; c < 4; ++c) {
            int col = h * HD + c * 16 + m;
            Op[(size_t)(b * N1 + row) * EMB + col] = f2bf(o[c][j] * inv);
        }
    }
}

// ---------------------------------------------------------------------------
// LayerNorm over rows of 1024; one 256-thread block per row.
__global__ void layernorm_kernel(const float* __restrict__ X,
                                 const float* __restrict__ gamma,
                                 const float* __restrict__ beta,
                                 float* __restrict__ out) {
    __shared__ float red[256];
    int row = blockIdx.x;
    int t   = threadIdx.x;
    const float* x = X + (size_t)row * EMB;

    float s = 0.0f, s2 = 0.0f;
    for (int i = t; i < EMB; i += 256) {
        float v = x[i];
        s += v; s2 += v * v;
    }
    red[t] = s; __syncthreads();
    for (int off = 128; off > 0; off >>= 1) {
        if (t < off) red[t] += red[t + off];
        __syncthreads();
    }
    float mean = red[0] / (float)EMB;
    __syncthreads();
    red[t] = s2; __syncthreads();
    for (int off = 128; off > 0; off >>= 1) {
        if (t < off) red[t] += red[t + off];
        __syncthreads();
    }
    float var = red[0] / (float)EMB - mean * mean;
    float inv = rsqrtf(var + 1e-5f);
    for (int i = t; i < EMB; i += 256) {
        float v = (x[i] - mean) * inv;
        out[(size_t)row * EMB + i] = v * gamma[i] + beta[i];
    }
}

// ---------------------------------------------------------------------------
extern "C" void kernel_launch(void* const* d_in, const int* in_sizes, int n_in,
                              void* d_out, int out_size, void* d_ws, size_t ws_size,
                              hipStream_t stream) {
    (void)in_sizes; (void)n_in; (void)out_size; (void)ws_size;

    const float* query = (const float*)d_in[0];
    const float* key   = (const float*)d_in[1];
    const float* value = (const float*)d_in[2];
    const float* Wq    = (const float*)d_in[3];
    const float* bq    = (const float*)d_in[4];
    const float* Wk    = (const float*)d_in[5];
    const float* bk    = (const float*)d_in[6];
    const float* Wv    = (const float*)d_in[7];
    const float* bv    = (const float*)d_in[8];
    const float* Wo    = (const float*)d_in[9];
    const float* bo    = (const float*)d_in[10];
    const float* gamma = (const float*)d_in[11];
    const float* beta  = (const float*)d_in[12];
    float* out = (float*)d_out;

    const int Mrows = BB * N1;                 // 4096
    char* ws = (char*)d_ws;
    size_t off = 0;
    auto alloc = [&](size_t bytes) {
        size_t r = off;
        off += (bytes + 255) & ~(size_t)255;
        return r;
    };
    bf16* q_bf   = (bf16*)(ws + alloc((size_t)Mrows * QD * 2));
    bf16* k_bf   = (bf16*)(ws + alloc((size_t)BB * N2 * QD * 2));
    bf16* v_bf   = (bf16*)(ws + alloc((size_t)BB * N2 * QD * 2));
    bf16* Wqt    = (bf16*)(ws + alloc((size_t)EMB * QD * 2));
    bf16* Wkt    = (bf16*)(ws + alloc((size_t)EMB * QD * 2));
    bf16* Wvt    = (bf16*)(ws + alloc((size_t)EMB * QD * 2));
    bf16* Wot    = (bf16*)(ws + alloc((size_t)EMB * EMB * 2));
    bf16* Qproj  = (bf16*)(ws + alloc((size_t)Mrows * EMB * 2));
    bf16* Kproj  = (bf16*)(ws + alloc((size_t)BB * N2 * EMB * 2));
    bf16* Vproj  = (bf16*)(ws + alloc((size_t)BB * N2 * EMB * 2));
    bf16* Vtr    = (bf16*)(ws + alloc((size_t)BB * NH * HD * N2 * 2));
    bf16* attnO  = (bf16*)(ws + alloc((size_t)Mrows * EMB * 2));
    float* preLN = (float*)(ws + alloc((size_t)Mrows * EMB * 4));

    // 1) activations f32 -> bf16
    {
        int n = Mrows * QD;
        int blocks = (n + 255) / 256;
        cvt_f32_bf16_kernel<<<blocks, 256, 0, stream>>>(query, q_bf, n);
        cvt_f32_bf16_kernel<<<blocks, 256, 0, stream>>>(key,   k_bf, n);
        cvt_f32_bf16_kernel<<<blocks, 256, 0, stream>>>(value, v_bf, n);
    }
    // 2) weights: transpose + convert (Wt[N][K], K contiguous for B-fragments)
    {
        int n = QD * EMB;
        int blocks = (n + 255) / 256;
        cvt_transpose_kernel<<<blocks, 256, 0, stream>>>(Wq, Wqt, QD, EMB);
        cvt_transpose_kernel<<<blocks, 256, 0, stream>>>(Wk, Wkt, QD, EMB);
        cvt_transpose_kernel<<<blocks, 256, 0, stream>>>(Wv, Wvt, QD, EMB);
        int n2 = EMB * EMB;
        cvt_transpose_kernel<<<(n2 + 255) / 256, 256, 0, stream>>>(Wo, Wot, EMB, EMB);
    }
    // 3) Q/K/V projections (WMMA GEMM, bias fused, bf16 out)
    {
        dim3 grid(EMB / 64, Mrows / 32);
        gemm_bias_wmma_kernel<bf16><<<grid, 32, 0, stream>>>(
            q_bf, Wqt, bq, Qproj, Mrows, EMB, QD);
        gemm_bias_wmma_kernel<bf16><<<grid, 32, 0, stream>>>(
            k_bf, Wkt, bk, Kproj, Mrows, EMB, QD);
        gemm_bias_wmma_kernel<bf16><<<grid, 32, 0, stream>>>(
            v_bf, Wvt, bv, Vproj, Mrows, EMB, QD);
    }
    // 4) V -> per-head transposed layout [b,h,d][n2]
    {
        int n = BB * N2 * EMB;
        v_transpose_kernel<<<(n + 255) / 256, 256, 0, stream>>>(Vproj, Vtr);
    }
    // 5) flash attention
    {
        dim3 grid(N1 / 16, NH, BB);
        attn_flash_kernel<<<grid, 32, 0, stream>>>(Qproj, Kproj, Vtr, attnO);
    }
    // 6) output projection (f32 out for LayerNorm)
    {
        dim3 grid(EMB / 64, Mrows / 32);
        gemm_bias_wmma_kernel<float><<<grid, 32, 0, stream>>>(
            attnO, Wot, bo, preLN, Mrows, EMB, EMB);
    }
    // 7) LayerNorm
    layernorm_kernel<<<Mrows, 256, 0, stream>>>(preLN, gamma, beta, out);
}
